// AttnBlock_66391604462051
// MI455X (gfx1250) — matrix operand order repaired and checked by reading, not compiled, not run on hardware
//
#include <hip/hip_runtime.h>

typedef __attribute__((ext_vector_type(16))) __bf16 v16bf;
typedef __attribute__((ext_vector_type(8)))  __bf16 v8bf;
typedef __attribute__((ext_vector_type(8)))  float  v8f;

#define WMMA_BF16(a, b, c) \
  __builtin_amdgcn_wmma_f32_16x16x32_bf16(false, (a), false, (b), (short)0, (c), false, false)

// ---------------------------------------------------------------------------
// WMMA fragment loaders (wave32 layouts per CDNA5 ISA 7.12.2)
// A 16x32 bf16: lane L -> row L&15; lanes 0-15 hold K {0..7,16..23},
//               lanes 16-31 hold K {8..15,24..31} (2 packed per VGPR).
// B 32x16 bf16: lane L -> col L&15; lanes 0-15 hold K {0..7,16..23},
//               lanes 16-31 hold K {8..15,24..31}.
// C/D 16x16 f32: VGPR j -> row j + 8*(L>>4), col L&15.
// ---------------------------------------------------------------------------
static __device__ __forceinline__ v16bf load_afrag(const __bf16* p0, int ld) {
  const int lane = threadIdx.x & 31;
  const __bf16* p = p0 + (size_t)(lane & 15) * ld + ((lane >> 4) << 3);
  union { v16bf v; v8bf h[2]; } u;
  u.h[0] = *(const v8bf*)(p);        // K = 8h .. 8h+7
  u.h[1] = *(const v8bf*)(p + 16);   // K = 16+8h .. 16+8h+7
  return u.v;
}

static __device__ __forceinline__ v16bf load_bfrag(const __bf16* p0, int ld) {
  const int lane = threadIdx.x & 31;
  const __bf16* p = p0 + (size_t)(lane & 15) * ld + ((lane >> 4) << 3);
  union { v16bf v; v8bf h[2]; } u;
  u.h[0] = *(const v8bf*)(p);        // K = 8h .. 8h+7
  u.h[1] = *(const v8bf*)(p + 16);   // K = 16+8h .. 16+8h+7
  return u.v;
}

// A fragment sourced from f32 LDS (softmax probabilities), convert to bf16.
static __device__ __forceinline__ v16bf load_afrag_f32(const float* p0, int ld) {
  const int lane = threadIdx.x & 31;
  const float* p = p0 + (lane & 15) * ld + ((lane >> 4) << 3);
  v16bf f;
#pragma unroll
  for (int i = 0; i < 8; ++i) {
    f[i]     = (__bf16)p[i];
    f[i + 8] = (__bf16)p[16 + i];
  }
  return f;
}

// CDNA5 async copy: 16B per lane, global -> LDS, tracked by ASYNCcnt.
static __device__ __forceinline__ void async_g2l_b128(unsigned lds_off,
                                                      const __bf16* gaddr) {
  asm volatile("global_load_async_to_lds_b128 %0, %1, off"
               :: "v"(lds_off), "v"(gaddr) : "memory");
}
static __device__ __forceinline__ void wait_async0() {
  asm volatile("s_wait_asynccnt 0x0" ::: "memory");
}

// ---------------------------------------------------------------------------
// Kernel 1: convert 4 weight matrices f32 [K=512][N=512] -> bf16 WT [N][K]
// ---------------------------------------------------------------------------
__global__ __launch_bounds__(256) void wconv_kernel(
    const float* __restrict__ Wq, const float* __restrict__ Wk,
    const float* __restrict__ Wv, const float* __restrict__ Wo,
    __bf16* __restrict__ T) {
  int id = blockIdx.x * 256 + threadIdx.x;      // 4 * 512 * 512 total
  int w  = id >> 18;                             // which matrix
  int r  = id & 262143;
  int k  = r >> 9;
  int n  = r & 511;
  const float* W = (w == 0) ? Wq : (w == 1) ? Wk : (w == 2) ? Wv : Wo;
  T[(size_t)w * 262144 + (size_t)n * 512 + k] = (__bf16)W[(size_t)k * 512 + n];
}

// ---------------------------------------------------------------------------
// Kernel 2: GroupNorm over (H,W,16ch) per (batch,group); emit h in bf16.
// ---------------------------------------------------------------------------
__global__ __launch_bounds__(256) void gn_kernel(
    const float* __restrict__ x, const float* __restrict__ gsc,
    const float* __restrict__ gbi, __bf16* __restrict__ h) {
  const int b = blockIdx.x >> 5;
  const int g = blockIdx.x & 31;
  __shared__ float r1[256], r2[256];
  const size_t base = (size_t)b * 1024 * 512 + (size_t)g * 16;

  float s = 0.f, sq = 0.f;
  for (int e = threadIdx.x; e < 16384; e += 256) {
    int n = e >> 4, c = e & 15;
    float v = x[base + (size_t)n * 512 + c];
    s += v; sq += v * v;
  }
  r1[threadIdx.x] = s; r2[threadIdx.x] = sq;
  __syncthreads();
  for (int off = 128; off > 0; off >>= 1) {
    if (threadIdx.x < off) {
      r1[threadIdx.x] += r1[threadIdx.x + off];
      r2[threadIdx.x] += r2[threadIdx.x + off];
    }
    __syncthreads();
  }
  const float mean = r1[0] * (1.0f / 16384.0f);
  const float var  = r2[0] * (1.0f / 16384.0f) - mean * mean;
  const float rstd = rsqrtf(var + 1e-6f);

  for (int e = threadIdx.x; e < 16384; e += 256) {
    int n = e >> 4, c = e & 15;
    int ch = g * 16 + c;
    float v = (x[base + (size_t)n * 512 + c] - mean) * rstd * gsc[ch] + gbi[ch];
    h[base + (size_t)n * 512 + c] = (__bf16)v;
  }
}

// ---------------------------------------------------------------------------
// Kernel 3: bf16 GEMM  out = A[M=32768,K=512] * W[K,N=512] + bias.
// BT is W transposed bf16 [N][K]. Block tile 128x128, 8 waves in 4(M)x2(N).
// Weight tile (shared by 4 waves) is double-buffer staged into LDS with
// CDNA5 async copies (global_load_async_to_lds_b128 / s_wait_asynccnt);
// A slabs are prefetched one K-step ahead (global_prefetch_b8).
//  MODE 0: bf16 out, (acc+bias)*scale   (Q, attention scale folded)
//  MODE 1: bf16 out, acc+bias           (K)
//  MODE 2: bf16 out transposed per batch vT[b][n][key]  (V)
//  MODE 3: f32 out = acc + bias + resid (O-proj + residual)
// ---------------------------------------------------------------------------
#define BLD 40   // padded LDS row stride in bf16 elems (80B -> conflict-free-ish)

template <int MODE>
__global__ __launch_bounds__(256) void gemm_proj_kernel(
    const __bf16* __restrict__ A, const __bf16* __restrict__ BT,
    const float* __restrict__ bias, const float* __restrict__ resid,
    void* __restrict__ out, float scale) {
  __shared__ __bf16 Bs[2][128 * BLD];

  const int tid  = threadIdx.x;
  const int wave = tid >> 5;
  const int lane = tid & 31;
  const int wm = wave & 3;          // 0..3 (M)
  const int wn = wave >> 2;         // 0..1 (N)
  const int mbase = blockIdx.x * 128 + wm * 32;
  const int nblk  = blockIdx.y * 128;
  const int nbase = nblk + wn * 64;

  // Stage the 128x32 weight tile for K-step ks into LDS buffer `buf`.
  // 512 chunks of 16B; 2 per thread. chunk c: row=c>>2, col-chunk=(c&3)*8.
  auto stage = [&](int buf, int ks) {
    const int ko = ks * 32;
#pragma unroll
    for (int u = 0; u < 2; ++u) {
      const int c   = tid * 2 + u;
      const int row = c >> 2;
      const int cc  = (c & 3) * 8;
      const __bf16* g = BT + (size_t)(nblk + row) * 512 + ko + cc;
      unsigned loff = (unsigned)(size_t)&Bs[buf][row * BLD + cc];
      async_g2l_b128(loff, g);
    }
  };

  v8f acc[2][4];
#pragma unroll
  for (int s = 0; s < 2; ++s)
#pragma unroll
    for (int nt = 0; nt < 4; ++nt) acc[s][nt] = (v8f)0.0f;

  stage(0, 0);
  wait_async0();
  __syncthreads();

  for (int ks = 0; ks < 16; ++ks) {
    const int cur = ks & 1;
    const int ko  = ks * 32;
    if (ks + 1 < 16) {
      stage(1 - cur, ks + 1);   // buffer 1-cur last read at ks-1, behind barrier
      // prefetch next A slab: one line per lane covers this wave's 32 rows
      __builtin_prefetch(A + (size_t)(mbase + lane) * 512 + (ks + 1) * 32, 0, 0);
    }
    v16bf a0 = load_afrag(A + (size_t)mbase * 512 + ko, 512);
    v16bf a1 = load_afrag(A + (size_t)(mbase + 16) * 512 + ko, 512);
#pragma unroll
    for (int nt = 0; nt < 4; ++nt) {
      v16bf bfr = load_bfrag(&Bs[cur][(wn * 64 + nt * 16) * BLD], BLD);
      acc[0][nt] = WMMA_BF16(a0, bfr, acc[0][nt]);
      acc[1][nt] = WMMA_BF16(a1, bfr, acc[1][nt]);
    }
    if (ks + 1 < 16) {
      wait_async0();            // own async stores to LDS complete
      __syncthreads();          // all waves' staging visible
    }
  }

  const int col = lane & 15;
  const int rowoff = (lane >> 4) * 8;
#pragma unroll
  for (int s = 0; s < 2; ++s) {
#pragma unroll
    for (int nt = 0; nt < 4; ++nt) {
      const int n = nbase + nt * 16 + col;
      const float bv = bias[n];
#pragma unroll
      for (int j = 0; j < 8; ++j) {
        const int m = mbase + s * 16 + rowoff + j;
        float val = acc[s][nt][j] + bv;
        if constexpr (MODE == 0) {
          ((__bf16*)out)[(size_t)m * 512 + n] = (__bf16)(val * scale);
        } else if constexpr (MODE == 1) {
          ((__bf16*)out)[(size_t)m * 512 + n] = (__bf16)val;
        } else if constexpr (MODE == 2) {
          const int b = m >> 10, key = m & 1023;
          ((__bf16*)out)[((size_t)b * 512 + n) * 1024 + key] = (__bf16)val;
        } else {
          ((float*)out)[(size_t)m * 512 + n] = val + resid[(size_t)m * 512 + n];
        }
      }
    }
  }
}

// ---------------------------------------------------------------------------
// Kernel 4: attention for one (batch, 16-query tile).
// Phase 1: S = q @ k^T  (scale pre-folded into q), 8 waves x 128 keys each.
// Phase 2: row softmax in LDS (2 rows per wave, wave32 shfl reductions).
// Phase 3: O = P @ V, 8 waves x 64 output channels each (vT layout).
// Dynamic LDS: 16 rows x 1040 f32 = 66560 B.
// ---------------------------------------------------------------------------
#define SLD 1040

__global__ __launch_bounds__(256) void attn_kernel(
    const __bf16* __restrict__ q, const __bf16* __restrict__ k,
    const __bf16* __restrict__ vT, __bf16* __restrict__ attn) {
  extern __shared__ float S[];  // [16][SLD]
  const int b = blockIdx.x >> 6;       // 64 query tiles per batch
  const int qt = blockIdx.x & 63;
  const int wave = threadIdx.x >> 5;
  const int lane = threadIdx.x & 31;
  const size_t qoff = ((size_t)b * 1024 + (size_t)qt * 16) * 512;
  const size_t koff = (size_t)b * 1024 * 512;

  // ---- Phase 1: scores into LDS --------------------------------------
  v8f sc[8];
#pragma unroll
  for (int kt = 0; kt < 8; ++kt) sc[kt] = (v8f)0.0f;

  for (int ks = 0; ks < 16; ++ks) {
    const int ko = ks * 32;
    v16bf aq = load_afrag(q + qoff + ko, 512);
#pragma unroll
    for (int kt = 0; kt < 8; ++kt) {
      v16bf bk = load_bfrag(k + koff + (size_t)(wave * 128 + kt * 16) * 512 + ko, 512);
      sc[kt] = WMMA_BF16(aq, bk, sc[kt]);
    }
  }
  {
    const int col = lane & 15, rowoff = (lane >> 4) * 8;
#pragma unroll
    for (int kt = 0; kt < 8; ++kt)
#pragma unroll
      for (int j = 0; j < 8; ++j)
        S[(rowoff + j) * SLD + wave * 128 + kt * 16 + col] = sc[kt][j];
  }
  __syncthreads();

  // ---- Phase 2: softmax over 1024 keys, 2 rows per wave --------------
#pragma unroll
  for (int rr = 0; rr < 2; ++rr) {
    float* row = S + (wave * 2 + rr) * SLD;
    float m = -3.4e38f;
    for (int i = lane; i < 1024; i += 32) m = fmaxf(m, row[i]);
#pragma unroll
    for (int off = 16; off > 0; off >>= 1) m = fmaxf(m, __shfl_xor(m, off, 32));
    float sum = 0.f;
    for (int i = lane; i < 1024; i += 32) {
      float e = __expf(row[i] - m);
      row[i] = e;
      sum += e;
    }
#pragma unroll
    for (int off = 16; off > 0; off >>= 1) sum += __shfl_xor(sum, off, 32);
    const float inv = 1.0f / sum;
    for (int i = lane; i < 1024; i += 32) row[i] *= inv;
  }
  __syncthreads();

  // ---- Phase 3: O = P @ V, wave owns 64 channels ---------------------
  v8f o[4];
#pragma unroll
  for (int dt = 0; dt < 4; ++dt) o[dt] = (v8f)0.0f;

  for (int kk = 0; kk < 32; ++kk) {
    v16bf pa = load_afrag_f32(S + kk * 32, SLD);
#pragma unroll
    for (int dt = 0; dt < 4; ++dt) {
      v16bf bv = load_bfrag(
          vT + ((size_t)b * 512 + (size_t)(wave * 64 + dt * 16)) * 1024 + kk * 32, 1024);
      o[dt] = WMMA_BF16(pa, bv, o[dt]);
    }
  }
  {
    const int col = lane & 15, rowoff = (lane >> 4) * 8;
#pragma unroll
    for (int dt = 0; dt < 4; ++dt)
#pragma unroll
      for (int j = 0; j < 8; ++j)
        attn[((size_t)b * 1024 + (size_t)qt * 16 + rowoff + j) * 512 +
             wave * 64 + dt * 16 + col] = (__bf16)o[dt][j];
  }
}

// ---------------------------------------------------------------------------
// Launcher
// ---------------------------------------------------------------------------
extern "C" void kernel_launch(void* const* d_in, const int* in_sizes, int n_in,
                              void* d_out, int out_size, void* d_ws, size_t ws_size,
                              hipStream_t stream) {
  const float* x   = (const float*)d_in[0];
  const float* gsc = (const float*)d_in[1];
  const float* gbi = (const float*)d_in[2];
  const float* Wq  = (const float*)d_in[3];
  const float* bq  = (const float*)d_in[4];
  const float* Wk  = (const float*)d_in[5];
  const float* bk  = (const float*)d_in[6];
  const float* Wv  = (const float*)d_in[7];
  const float* bv  = (const float*)d_in[8];
  const float* Wo  = (const float*)d_in[9];
  const float* bo  = (const float*)d_in[10];
  float* out = (float*)d_out;

  // Workspace layout (B=32, N=1024, C=512; M = 32768)
  const size_t MC = (size_t)32768 * 512;       // elements per bf16 tensor
  char* ws = (char*)d_ws;
  __bf16* h    = (__bf16*)(ws);                      // 33.5 MB (reused for attn out)
  __bf16* qb   = (__bf16*)(ws + 1 * MC * 2);
  __bf16* kb   = (__bf16*)(ws + 2 * MC * 2);
  __bf16* vTb  = (__bf16*)(ws + 3 * MC * 2);
  __bf16* WT   = (__bf16*)(ws + 4 * MC * 2);         // 4 x 512x512 bf16
  __bf16* attn = h;                                  // h dead after V GEMM

  const float qscale = 0.04419417382415922f;         // 512^-0.5

  // 1) weights -> bf16 transposed
  wconv_kernel<<<4096, 256, 0, stream>>>(Wq, Wk, Wv, Wo, WT);
  // 2) GroupNorm -> h (bf16)
  gn_kernel<<<1024, 256, 0, stream>>>(x, gsc, gbi, h);
  // 3) Q/K/V projections
  dim3 gg(256, 4);
  gemm_proj_kernel<0><<<gg, 256, 0, stream>>>(h, WT + 0 * 262144, bq, nullptr, qb, qscale);
  gemm_proj_kernel<1><<<gg, 256, 0, stream>>>(h, WT + 1 * 262144, bk, nullptr, kb, 1.0f);
  gemm_proj_kernel<2><<<gg, 256, 0, stream>>>(h, WT + 2 * 262144, bv, nullptr, vTb, 1.0f);
  // 4) attention (scores + softmax + PV); writes attn (= h buffer)
  attn_kernel<<<2048, 256, 16 * SLD * sizeof(float), stream>>>(qb, kb, vTb, attn);
  // 5) output projection + bias + residual -> f32 out
  gemm_proj_kernel<3><<<gg, 256, 0, stream>>>(attn, WT + 3 * 262144, bo, x, out, 1.0f);
}
